// TemporalEncoder_73186242724016
// MI455X (gfx1250) — matrix operand order repaired and checked by reading, not compiled, not run on hardware
//
#include <hip/hip_runtime.h>
#include <hip/hip_bf16.h>

// ---------------------------------------------------------------------------
// ConvGRU (2-layer, T=16) on gfx1250 via implicit-GEMM bf16 WMMA.
// Combined-input layout per cell: channels [0..31] = cell's own hidden state,
// [32..63] = "extra" input (x for layer0 packed as {x,0,...}, h0 for layer1).
// Weights pre-packed to the exact per-lane V_WMMA B-fragment layout.
// ---------------------------------------------------------------------------

typedef __attribute__((ext_vector_type(16))) __bf16 v16bf;
typedef __attribute__((ext_vector_type(8)))  float  v8f;

#define Bn 8
#define Tn 16
#define Hn 128
#define Wn 128
#define Cn 32          // hidden channels
#define TILE_COLS 66   // 64-pixel strip + 2 halo
#define SMEM_ELEMS (3 * TILE_COLS * 64)

struct Frag32 { uint4 lo; uint4 hi; };   // 32 bytes == v16bf

__device__ __forceinline__ unsigned short f2bf(float f) {
  unsigned int u = __float_as_uint(f);
  u += 0x7fffu + ((u >> 16) & 1u);       // round-to-nearest-even
  return (unsigned short)(u >> 16);
}
__device__ __forceinline__ float bf2f(unsigned short h) {
  return __uint_as_float(((unsigned int)h) << 16);
}

// Stage [3 rows][66 cols][64 ch] bf16 input tile (with SAME-padding zeros).
__device__ __forceinline__ void stage_tile(
    unsigned short* __restrict__ smem,
    const unsigned short* __restrict__ hid,     // 32-ch [B,H,W,32] bf16
    const unsigned short* __restrict__ extra,   // x (1ch) or 32-ch tensor
    int extraIsX, long extraBStride,
    int b, int y, int xb, int tid) {
  for (int p = tid; p < 3 * TILE_COLS; p += 128) {
    int row = p / TILE_COLS, col = p % TILE_COLS;
    int gy = y + row - 1;
    int gx = xb + col - 1;
    bool valid = (gy >= 0) && (gy < Hn) && (gx >= 0) && (gx < Wn);
    uint4* dst = (uint4*)&smem[p * 64];
    uint4 z4; z4.x = z4.y = z4.z = z4.w = 0u;
    if (valid) {
      const uint4* s0 =
          (const uint4*)(hid + (((size_t)b * Hn + gy) * Wn + gx) * Cn);
      dst[0] = s0[0]; dst[1] = s0[1]; dst[2] = s0[2]; dst[3] = s0[3];
      if (extraIsX) {
        unsigned short xv =
            extra[(size_t)b * extraBStride + (size_t)gy * Wn + gx];
        uint4 u; u.x = (unsigned int)xv; u.y = u.z = u.w = 0u;
        dst[4] = u; dst[5] = z4; dst[6] = z4; dst[7] = z4;
      } else {
        const uint4* s1 =
            (const uint4*)(extra + (((size_t)b * Hn + gy) * Wn + gx) * Cn);
        dst[4] = s1[0]; dst[5] = s1[1]; dst[6] = s1[2]; dst[7] = s1[3];
      }
    } else {
      #pragma unroll
      for (int i = 0; i < 8; ++i) dst[i] = z4;
    }
  }
}

// 16-pixel x (NT*16)-outchannel implicit-GEMM conv: K = 9 taps x 64 channels.
template <int NT>
__device__ __forceinline__ void conv_gemm(
    const unsigned short* __restrict__ smem,
    const unsigned short* __restrict__ wp,   // packed B fragments
    int lane, int wave, v8f acc[NT]) {
  const int m  = lane & 15;            // A: M index (pixel in tile)
  const int kb = (lane >> 4) << 3;     // A: K base (0 or 8) per lane-half
  #pragma unroll
  for (int i = 0; i < NT; ++i)
    #pragma unroll
    for (int j = 0; j < 8; ++j) acc[i][j] = 0.f;

  for (int tap = 0; tap < 9; ++tap) {
    const int ky = tap / 3, kx = tap % 3;
    #pragma unroll
    for (int ch = 0; ch < 2; ++ch) {
      const int col  = wave * 16 + m + kx;
      const int base = (ky * TILE_COLS + col) * 64 + ch * 32 + kb;
      Frag32 fa;
      fa.lo = *(const uint4*)&smem[base];        // K = kb .. kb+7
      fa.hi = *(const uint4*)&smem[base + 16];   // K = kb+16 .. kb+23
      v16bf a = __builtin_bit_cast(v16bf, fa);

      const unsigned short* wl =
          wp + ((size_t)(tap * 2 + ch) * NT) * 512 + lane * 16;
      __builtin_prefetch(wl + NT * 512, 0, 1);   // next (tap,chunk) weights
      #pragma unroll
      for (int nt = 0; nt < NT; ++nt) {
        Frag32 fb;
        fb.lo = *(const uint4*)(wl + (size_t)nt * 512);
        fb.hi = *(const uint4*)(wl + (size_t)nt * 512 + 8);
        v16bf bm = __builtin_bit_cast(v16bf, fb);
        acc[nt] = __builtin_amdgcn_wmma_f32_16x16x32_bf16(
            false, a, false, bm, (short)0, acc[nt], false, false);
      }
    }
  }
}

// Gates: sigmoid(conv([h|extra])) -> writes r*h (bf16) and z (f32).
__global__ __launch_bounds__(128) void convgru_gates_kernel(
    const unsigned short* __restrict__ hid,
    const unsigned short* __restrict__ extra, int extraIsX, long extraBStride,
    const unsigned short* __restrict__ wp, const float* __restrict__ gbias,
    unsigned short* __restrict__ rh, float* __restrict__ zbuf) {
  __shared__ __align__(16) unsigned short smem[SMEM_ELEMS];
  const int tid = threadIdx.x;
  const int strip = blockIdx.x, y = blockIdx.y, b = blockIdx.z;
  const int xb = strip * 64;
  stage_tile(smem, hid, extra, extraIsX, extraBStride, b, y, xb, tid);
  __syncthreads();

  const int wave = tid >> 5, lane = tid & 31;
  v8f acc[4];
  conv_gemm<4>(smem, wp, lane, wave, acc);

  const int n  = lane & 15;
  const int mb = (lane >> 4) * 8;
  #pragma unroll
  for (int nt = 0; nt < 4; ++nt) {
    const int co = nt * 16 + n;
    const float bias = gbias[co];
    #pragma unroll
    for (int vg = 0; vg < 8; ++vg) {
      const int m  = vg + mb;
      const int gx = xb + wave * 16 + m;
      const float s = 1.f / (1.f + __expf(-(acc[nt][vg] + bias)));
      const size_t pix = ((size_t)b * Hn + y) * Wn + gx;
      if (co < Cn) {   // reset gate -> r*h (h read back from staged chunk0)
        float h = bf2f(smem[(TILE_COLS + wave * 16 + m + 1) * 64 + co]);
        rh[pix * Cn + co] = f2bf(s * h);
      } else {         // update gate z
        zbuf[pix * Cn + (co - Cn)] = s;
      }
    }
  }
}

// Candidate: n = tanh(conv([rh|extra])); h_new = (1-z)h + z n.
__global__ __launch_bounds__(128) void convgru_cand_kernel(
    const unsigned short* __restrict__ rh,
    const unsigned short* __restrict__ extra, int extraIsX, long extraBStride,
    const unsigned short* __restrict__ wp, const float* __restrict__ cbias,
    const unsigned short* __restrict__ hprev, const float* __restrict__ zbuf,
    unsigned short* __restrict__ hnew, float* __restrict__ outp, int writeOut) {
  __shared__ __align__(16) unsigned short smem[SMEM_ELEMS];
  const int tid = threadIdx.x;
  const int strip = blockIdx.x, y = blockIdx.y, b = blockIdx.z;
  const int xb = strip * 64;
  stage_tile(smem, rh, extra, extraIsX, extraBStride, b, y, xb, tid);
  __syncthreads();

  const int wave = tid >> 5, lane = tid & 31;
  v8f acc[2];
  conv_gemm<2>(smem, wp, lane, wave, acc);

  const int n  = lane & 15;
  const int mb = (lane >> 4) * 8;
  #pragma unroll
  for (int nt = 0; nt < 2; ++nt) {
    const int co = nt * 16 + n;
    const float bias = cbias[co];
    #pragma unroll
    for (int vg = 0; vg < 8; ++vg) {
      const int m  = vg + mb;
      const int gx = xb + wave * 16 + m;
      const float nv = tanhf(acc[nt][vg] + bias);
      const size_t idx = (((size_t)b * Hn + y) * Wn + gx) * Cn + co;
      const float hp = bf2f(hprev[idx]);
      const float z  = zbuf[idx];
      const float hn = (1.f - z) * hp + z * nv;
      hnew[idx] = f2bf(hn);
      if (writeOut)   // final h1 in fp32 NCHW
        outp[(((size_t)b * Cn + co) * Hn + y) * Wn + gx] = hn;
    }
  }
}

// fp32 -> bf16 elementwise pack of the input sequence.
__global__ void pack_x_kernel(const float* __restrict__ src,
                              unsigned short* __restrict__ dst, int n) {
  int i = blockIdx.x * blockDim.x + threadIdx.x;
  if (i < n) dst[i] = f2bf(src[i]);
}

// Pack OIHW fp32 conv weights into per-lane WMMA B-fragment order.
// Linear idx = ((tap*2+chunk)*NT + nt)*512 + lane*16 + e.
__global__ void pack_w_kernel(const float* __restrict__ w,
                              unsigned short* __restrict__ dst,
                              int NT, int layer0, int total) {
  int idx = blockIdx.x * blockDim.x + threadIdx.x;
  if (idx >= total) return;
  int e    = idx & 15;
  int lane = (idx >> 4) & 31;
  int q    = idx >> 9;
  int nt   = q % NT; q /= NT;
  int ch   = q % 2;
  int tap  = q / 2;
  int ky = tap / 3, kx = tap % 3;
  int co = nt * 16 + (lane & 15);
  int k  = (lane >> 4) * 8 + (e & 7) + ((e >> 3) << 4);   // K within chunk (B layout: table order, 8h+e / 16+8h+e-8)
  int c  = ch * 32 + k;            // combined channel 0..63
  float val = 0.f;
  if (layer0) {                    // combined = [h(orig 1..32) | x(orig 0) pad]
    const int Cin = 33;
    if (c < 32)      val = w[((co * Cin + (c + 1)) * 3 + ky) * 3 + kx];
    else if (c == 32) val = w[((co * Cin + 0) * 3 + ky) * 3 + kx];
  } else {                         // combined = [h1(orig 32..63) | h0(orig 0..31)]
    const int Cin = 64;
    int ci = (c < 32) ? (c + 32) : (c - 32);
    val = w[((co * Cin + ci) * 3 + ky) * 3 + kx];
  }
  dst[idx] = f2bf(val);
}

extern "C" void kernel_launch(void* const* d_in, const int* in_sizes, int n_in,
                              void* d_out, int out_size, void* d_ws, size_t ws_size,
                              hipStream_t stream) {
  (void)in_sizes; (void)n_in; (void)out_size; (void)ws_size;
  const float* seq = (const float*)d_in[0];
  const float* gw0 = (const float*)d_in[1];
  const float* gb0 = (const float*)d_in[2];
  const float* cw0 = (const float*)d_in[3];
  const float* cb0 = (const float*)d_in[4];
  const float* gw1 = (const float*)d_in[5];
  const float* gb1 = (const float*)d_in[6];
  const float* cw1 = (const float*)d_in[7];
  const float* cb1 = (const float*)d_in[8];
  float* out = (float*)d_out;

  const size_t HW    = (size_t)Hn * Wn;
  const size_t hElems = (size_t)Bn * HW * Cn;          // per state tensor
  const size_t hBytes = hElems * 2;

  char* base = (char*)d_ws;
  size_t off = 0;
  auto alloc = [&](size_t bytes) -> void* {
    void* p = base + off;
    off = (off + bytes + 255) & ~(size_t)255;
    return p;
  };
  unsigned short* xbf   = (unsigned short*)alloc((size_t)Bn * Tn * HW * 2);
  unsigned short* h0b[2] = {(unsigned short*)alloc(hBytes),
                            (unsigned short*)alloc(hBytes)};
  unsigned short* h1b[2] = {(unsigned short*)alloc(hBytes),
                            (unsigned short*)alloc(hBytes)};
  unsigned short* rh    = (unsigned short*)alloc(hBytes);
  float*          zbuf  = (float*)alloc(hElems * 4);
  const int GW_TOT = 9 * 2 * 4 * 512;   // 36864 packed bf16 elems (gates)
  const int CW_TOT = 9 * 2 * 2 * 512;   // 18432 (candidate)
  unsigned short* wg0p = (unsigned short*)alloc((size_t)GW_TOT * 2);
  unsigned short* wc0p = (unsigned short*)alloc((size_t)CW_TOT * 2);
  unsigned short* wg1p = (unsigned short*)alloc((size_t)GW_TOT * 2);
  unsigned short* wc1p = (unsigned short*)alloc((size_t)CW_TOT * 2);

  // Zero initial hidden states (deterministic under graph replay).
  hipMemsetAsync(h0b[0], 0, hBytes, stream);
  hipMemsetAsync(h1b[0], 0, hBytes, stream);

  // One-time packs.
  const int NX = Bn * Tn * (int)HW;
  pack_x_kernel<<<(NX + 255) / 256, 256, 0, stream>>>(seq, xbf, NX);
  pack_w_kernel<<<(GW_TOT + 255) / 256, 256, 0, stream>>>(gw0, wg0p, 4, 1, GW_TOT);
  pack_w_kernel<<<(CW_TOT + 255) / 256, 256, 0, stream>>>(cw0, wc0p, 2, 1, CW_TOT);
  pack_w_kernel<<<(GW_TOT + 255) / 256, 256, 0, stream>>>(gw1, wg1p, 4, 0, GW_TOT);
  pack_w_kernel<<<(CW_TOT + 255) / 256, 256, 0, stream>>>(cw1, wc1p, 2, 0, CW_TOT);

  dim3 grid(Wn / 64, Hn, Bn);   // (2, 128, 8) blocks of 4 waves
  const long xBStride = (long)Tn * HW;     // batch stride of x in [B,T,H,W]
  const long hBStride = (long)HW * Cn;

  int c0 = 0, c1 = 0;
  for (int t = 0; t < Tn; ++t) {
    const unsigned short* xt = xbf + (size_t)t * HW;
    // ---- layer 0 ----
    convgru_gates_kernel<<<grid, 128, 0, stream>>>(
        h0b[c0], xt, 1, xBStride, wg0p, gb0, rh, zbuf);
    convgru_cand_kernel<<<grid, 128, 0, stream>>>(
        rh, xt, 1, xBStride, wc0p, cb0, h0b[c0], zbuf,
        h0b[1 - c0], out, 0);
    c0 ^= 1;
    // ---- layer 1 (extra input = freshly written h0) ----
    convgru_gates_kernel<<<grid, 128, 0, stream>>>(
        h1b[c1], h0b[c0], 0, hBStride, wg1p, gb1, rh, zbuf);
    convgru_cand_kernel<<<grid, 128, 0, stream>>>(
        rh, h0b[c0], 0, hBStride, wc1p, cb1, h1b[c1], zbuf,
        h1b[1 - c1], out, (t == Tn - 1) ? 1 : 0);
    c1 ^= 1;
  }
}
